// SimpleCompressedAttention_46050639348067
// MI455X (gfx1250) — matrix operand order repaired and checked by reading, not compiled, not run on hardware
//
#include <hip/hip_runtime.h>
#include <hip/hip_bf16.h>

#define NUM_HEADS 32
#define HEAD_DIM  128
#define LORA_RANK 512
#define BATCH     8
#define KV_LEN    4096
#define D_MODEL   4096
#define SCALE     0.08838834764831845f   /* 1/sqrt(128) */

typedef __attribute__((ext_vector_type(16))) __bf16         v16bf;
typedef __attribute__((ext_vector_type(8)))  float          v8f;
typedef __attribute__((ext_vector_type(4)))  unsigned int   u32x4;
typedef __attribute__((ext_vector_type(2)))  unsigned int   u32x2;
typedef __attribute__((ext_vector_type(4)))  float          f32x4;
typedef __attribute__((ext_vector_type(4)))  unsigned short u16x4;

union BF16Frag {
    unsigned short u[16];
    u32x4          q[2];
    v16bf          v;
};

__device__ __forceinline__ unsigned short f32_to_bf16(float f) {
    union { float f; unsigned int u; } c; c.f = f;
    unsigned int x = c.u;
    x += 0x7FFFu + ((x >> 16) & 1u);          // round-to-nearest-even
    return (unsigned short)(x >> 16);
}

// ---------------------------------------------------------------------------
// Workspace layout (bytes)
// ---------------------------------------------------------------------------
#define Q_OFF     0u          //  8*4096  f32  = 128 KB
#define QT_OFF    131072u     //  8*32*512 bf16 = 256 KB
#define SC_OFF    393216u     //  8*32*4096 f32 = 4 MB
#define PR_OFF    4587520u    //  8*32*4096 bf16 = 2 MB
#define CTX_OFF   6684672u    //  8*32*512  f32 = 512 KB
#define CKV16_OFF 7208960u    //  8*4096*512 bf16 (s-major) = 32 MB
#define CKVT_OFF  40763392u   //  8*512*4096 bf16 (r-major) = 32 MB
#define WS_NEED_PRE 40763392u
#define WS_NEED_TR  74317824u

// ---------------------------------------------------------------------------
// K0a: single pass over ckv fp32: emit bf16 s-major copy AND bf16 r-major
//      (transposed) copy via 32x32 LDS tile. 64 MB read + 64 MB write, once.
// ---------------------------------------------------------------------------
__global__ void k0_cvt_tr(const float* __restrict__ ckv,
                          unsigned short* __restrict__ ckv16,
                          unsigned short* __restrict__ ckvT) {
    __shared__ unsigned short tile[32][34];          // pad: no bank conflicts
    int b  = blockIdx.z;
    int s0 = blockIdx.x * 32;
    int r0 = blockIdx.y * 32;
    int t  = threadIdx.x;                            // 256 threads
    int sl = t >> 3;                                 // 0..31
    int rl = (t & 7) * 4;                            // 0,4,...,28

    f32x4 f = *(const f32x4*)(ckv + ((size_t)b * KV_LEN + s0 + sl) * LORA_RANK + r0 + rl);
    unsigned short h0 = f32_to_bf16(f.x), h1 = f32_to_bf16(f.y),
                   h2 = f32_to_bf16(f.z), h3 = f32_to_bf16(f.w);

    u16x4 rowv = {h0, h1, h2, h3};
    *(u16x4*)(ckv16 + ((size_t)b * KV_LEN + s0 + sl) * LORA_RANK + r0 + rl) = rowv;

    tile[rl + 0][sl] = h0;
    tile[rl + 1][sl] = h1;
    tile[rl + 2][sl] = h2;
    tile[rl + 3][sl] = h3;
    __syncthreads();

    int rr = t >> 3;                                 // 0..31 (r within tile)
    int ss = (t & 7) * 4;                            // s within tile
    u16x4 colv = {tile[rr][ss], tile[rr][ss + 1], tile[rr][ss + 2], tile[rr][ss + 3]};
    *(u16x4*)(ckvT + ((size_t)b * LORA_RANK + r0 + rr) * KV_LEN + s0 + ss) = colv;
}

// K0b: mid path: s-major bf16 copy only
__global__ void k0_cvt(const float* __restrict__ ckv, unsigned int* __restrict__ ckv16) {
    int i = blockIdx.x * blockDim.x + threadIdx.x;      // one float4 -> one uint2
    f32x4 f = ((const f32x4*)ckv)[i];
    u32x2 o;
    o.x = (unsigned int)f32_to_bf16(f.x) | ((unsigned int)f32_to_bf16(f.y) << 16);
    o.y = (unsigned int)f32_to_bf16(f.z) | ((unsigned int)f32_to_bf16(f.w) << 16);
    ((u32x2*)ckv16)[i] = o;
}

// ---------------------------------------------------------------------------
// K1: q[b,o] = hidden[b,:] . Wq[o,:] + bq[o]   (Wq streamed from HBM once)
// ---------------------------------------------------------------------------
__global__ void k1_qproj(const float* __restrict__ hidden, const float* __restrict__ wq,
                         const float* __restrict__ bq, float* __restrict__ q) {
    int o = blockIdx.x * blockDim.x + threadIdx.x;
    if (o >= D_MODEL) return;
    float acc[BATCH];
#pragma unroll
    for (int b = 0; b < BATCH; ++b) acc[b] = 0.f;
    const float* wrow = wq + (size_t)o * D_MODEL;
#pragma unroll 4
    for (int i = 0; i < D_MODEL; ++i) {
        float w = wrow[i];
#pragma unroll
        for (int b = 0; b < BATCH; ++b) acc[b] += hidden[b * D_MODEL + i] * w;
    }
    float bias = bq[o];
#pragma unroll
    for (int b = 0; b < BATCH; ++b) q[b * D_MODEL + o] = acc[b] + bias;
}

// ---------------------------------------------------------------------------
// K2: absorbed query  q~[b,h,r] = SCALE * sum_d q[b,h,d] * Wk_h[d,r]
//     (k bias dropped: constant over s -> softmax invariant)
// ---------------------------------------------------------------------------
__global__ void k2_qabsorb(const float* __restrict__ q, const float* __restrict__ kvb_w,
                           unsigned short* __restrict__ qt) {
    int idx = blockIdx.x * blockDim.x + threadIdx.x;   // (h, r)
    int r = idx & (LORA_RANK - 1);
    int h = idx >> 9;
    if (h >= NUM_HEADS) return;
    float acc[BATCH];
#pragma unroll
    for (int b = 0; b < BATCH; ++b) acc[b] = 0.f;
    for (int d = 0; d < HEAD_DIM; ++d) {
        float w = kvb_w[(size_t)(h * 256 + d) * LORA_RANK + r];
#pragma unroll
        for (int b = 0; b < BATCH; ++b) acc[b] += q[b * D_MODEL + h * HEAD_DIM + d] * w;
    }
#pragma unroll
    for (int b = 0; b < BATCH; ++b)
        qt[((size_t)b * NUM_HEADS + h) * LORA_RANK + r] = f32_to_bf16(acc[b] * SCALE);
}

// ---------------------------------------------------------------------------
// K3a: scores = q~ @ ckv^T from s-major bf16; every fragment = 2x b128.
// ---------------------------------------------------------------------------
__global__ void k3_scores_pre(const unsigned short* __restrict__ qt,
                              const unsigned short* __restrict__ ckv16,
                              float* __restrict__ scores) {
    int b    = blockIdx.z;
    int lane = threadIdx.x & 31;
    int wave = threadIdx.x >> 5;
    int m0   = (wave & 1) * 16;
    int n0   = blockIdx.x * 64 + (wave >> 1) * 16;
    int l15  = lane & 15;
    int kaof = (lane & 16);        // A: +16 for hi half-wave
    int kbof = (lane & 16);        // B: +16 for hi half-wave

    const unsigned short* arow = qt + ((size_t)b * NUM_HEADS + m0 + l15) * LORA_RANK + kaof;
    const unsigned short* brow = ckv16 + ((size_t)b * KV_LEN + n0 + l15) * LORA_RANK + kbof;

    v8f c = {};
#pragma unroll 4
    for (int k0 = 0; k0 < LORA_RANK; k0 += 32) {
        __builtin_prefetch(brow + k0 + 256, 0, 1);     // global_prefetch_b8
        BF16Frag a, bm;
        a.q[0]  = *(const u32x4*)(arow + k0);          // K = k0+kaof + 0..7
        a.q[1]  = *(const u32x4*)(arow + k0 + 8);      // K = k0+kaof + 8..15
        bm.q[0] = *(const u32x4*)(brow + k0);          // K = k0+kbof + 0..7
        bm.q[1] = *(const u32x4*)(brow + k0 + 8);      // K = k0+kbof + 8..15
        c = __builtin_amdgcn_wmma_f32_16x16x32_bf16(false, a.v, false, bm.v,
                                                    (short)0, c, false, false);
    }
    float* out = scores + ((size_t)b * NUM_HEADS + m0 + ((lane >> 4) << 3)) * KV_LEN + n0 + l15;
#pragma unroll
    for (int i = 0; i < 8; ++i) out[(size_t)i * KV_LEN] = c[i];
}

// K3b: min path: convert B in-loop from fp32
__global__ void k3_scores_cvt(const unsigned short* __restrict__ qt, const float* __restrict__ ckv,
                              float* __restrict__ scores) {
    int b    = blockIdx.z;
    int lane = threadIdx.x & 31;
    int wave = threadIdx.x >> 5;
    int m0   = (wave & 1) * 16;
    int n0   = blockIdx.x * 64 + (wave >> 1) * 16;
    int l15  = lane & 15;
    int kaof = (lane & 16);
    int kbof = (lane & 16);

    const unsigned short* arow = qt + ((size_t)b * NUM_HEADS + m0 + l15) * LORA_RANK + kaof;
    const float*          brow = ckv + ((size_t)b * KV_LEN + n0 + l15) * LORA_RANK;

    v8f c = {};
    for (int k0 = 0; k0 < LORA_RANK; k0 += 32) {
        __builtin_prefetch(brow + k0 + 128, 0, 1);
        BF16Frag a, bm;
        a.q[0] = *(const u32x4*)(arow + k0);
        a.q[1] = *(const u32x4*)(arow + k0 + 8);
#pragma unroll
        for (int j = 0; j < 16; ++j) bm.u[j] = f32_to_bf16(brow[k0 + j + kbof]);
        c = __builtin_amdgcn_wmma_f32_16x16x32_bf16(false, a.v, false, bm.v,
                                                    (short)0, c, false, false);
    }
    float* out = scores + ((size_t)b * NUM_HEADS + m0 + ((lane >> 4) << 3)) * KV_LEN + n0 + l15;
#pragma unroll
    for (int i = 0; i < 8; ++i) out[(size_t)i * KV_LEN] = c[i];
}

// ---------------------------------------------------------------------------
// K4: row softmax over s (4096), one block per (b,h); probs stored as bf16.
// ---------------------------------------------------------------------------
__global__ void k4_softmax(const float* __restrict__ scores, unsigned short* __restrict__ probs) {
    __shared__ float red[256];
    int row = blockIdx.x;                       // 0..255 = (b,h)
    const float* s = scores + (size_t)row * KV_LEN;
    unsigned short* p = probs + (size_t)row * KV_LEN;
    int t = threadIdx.x;

    float m = -3.0e38f;
    for (int i = t; i < KV_LEN; i += 256) m = fmaxf(m, s[i]);
    red[t] = m; __syncthreads();
    for (int off = 128; off > 0; off >>= 1) {
        if (t < off) red[t] = fmaxf(red[t], red[t + off]);
        __syncthreads();
    }
    m = red[0]; __syncthreads();

    float l = 0.f;
    for (int i = t; i < KV_LEN; i += 256) l += __expf(s[i] - m);
    red[t] = l; __syncthreads();
    for (int off = 128; off > 0; off >>= 1) {
        if (t < off) red[t] += red[t + off];
        __syncthreads();
    }
    float inv = 1.0f / red[0];
    for (int i = t; i < KV_LEN; i += 256) p[i] = f32_to_bf16(__expf(s[i] - m) * inv);
}

// ---------------------------------------------------------------------------
// K5a: ctx = probs @ ckv from r-major (transposed) bf16: B fragment is
//      lane-row-contiguous -> 2x b128 per fragment, same shape as K3a.
// ---------------------------------------------------------------------------
__global__ void k5_ctx_tr(const unsigned short* __restrict__ probs,
                          const unsigned short* __restrict__ ckvT,
                          float* __restrict__ ctx) {
    int b    = blockIdx.z;
    int lane = threadIdx.x & 31;
    int wave = threadIdx.x >> 5;
    int m0   = (wave & 1) * 16;
    int n0   = (blockIdx.x * 4 + (wave >> 1)) * 16;
    int l15  = lane & 15;
    int kaof = (lane & 16);
    int kbof = (lane & 16);

    const unsigned short* arow = probs + ((size_t)b * NUM_HEADS + m0 + l15) * KV_LEN + kaof;
    const unsigned short* brow = ckvT + ((size_t)b * LORA_RANK + n0 + l15) * KV_LEN + kbof;

    v8f c = {};
#pragma unroll 4
    for (int k0 = 0; k0 < KV_LEN; k0 += 32) {
        __builtin_prefetch(brow + k0 + 256, 0, 1);
        BF16Frag a, bm;
        a.q[0]  = *(const u32x4*)(arow + k0);
        a.q[1]  = *(const u32x4*)(arow + k0 + 8);
        bm.q[0] = *(const u32x4*)(brow + k0);
        bm.q[1] = *(const u32x4*)(brow + k0 + 8);
        c = __builtin_amdgcn_wmma_f32_16x16x32_bf16(false, a.v, false, bm.v,
                                                    (short)0, c, false, false);
    }
    float* out = ctx + ((size_t)b * NUM_HEADS + m0 + ((lane >> 4) << 3)) * LORA_RANK + n0 + l15;
#pragma unroll
    for (int i = 0; i < 8; ++i) out[(size_t)i * LORA_RANK] = c[i];
}

// K5b: mid path: B column-gather from s-major bf16
__global__ void k5_ctx_pre(const unsigned short* __restrict__ probs,
                           const unsigned short* __restrict__ ckv16,
                           float* __restrict__ ctx) {
    int b    = blockIdx.z;
    int lane = threadIdx.x & 31;
    int wave = threadIdx.x >> 5;
    int m0   = (wave & 1) * 16;
    int n0   = (blockIdx.x * 4 + (wave >> 1)) * 16;
    int l15  = lane & 15;
    int kaof = (lane & 16);
    int kbof = (lane & 16);

    const unsigned short* arow = probs + ((size_t)b * NUM_HEADS + m0 + l15) * KV_LEN + kaof;
    const unsigned short* bcol = ckv16 + (size_t)b * KV_LEN * LORA_RANK + n0 + l15;

    v8f c = {};
#pragma unroll 2
    for (int k0 = 0; k0 < KV_LEN; k0 += 32) {
        BF16Frag a, bm;
        a.q[0] = *(const u32x4*)(arow + k0);
        a.q[1] = *(const u32x4*)(arow + k0 + 8);
#pragma unroll
        for (int j = 0; j < 16; ++j)
            bm.u[j] = bcol[(size_t)(k0 + j + kbof) * LORA_RANK];
        c = __builtin_amdgcn_wmma_f32_16x16x32_bf16(false, a.v, false, bm.v,
                                                    (short)0, c, false, false);
    }
    float* out = ctx + ((size_t)b * NUM_HEADS + m0 + ((lane >> 4) << 3)) * LORA_RANK + n0 + l15;
#pragma unroll
    for (int i = 0; i < 8; ++i) out[(size_t)i * LORA_RANK] = c[i];
}

// K5c: min path: convert B in-loop from fp32
__global__ void k5_ctx_cvt(const unsigned short* __restrict__ probs, const float* __restrict__ ckv,
                           float* __restrict__ ctx) {
    int b    = blockIdx.z;
    int lane = threadIdx.x & 31;
    int wave = threadIdx.x >> 5;
    int m0   = (wave & 1) * 16;
    int n0   = (blockIdx.x * 4 + (wave >> 1)) * 16;
    int l15  = lane & 15;
    int kaof = (lane & 16);
    int kbof = (lane & 16);

    const unsigned short* arow = probs + ((size_t)b * NUM_HEADS + m0 + l15) * KV_LEN + kaof;
    const float*          bcol = ckv + (size_t)b * KV_LEN * LORA_RANK + n0 + l15;

    v8f c = {};
    for (int k0 = 0; k0 < KV_LEN; k0 += 32) {
        BF16Frag a, bm;
        a.q[0] = *(const u32x4*)(arow + k0);
        a.q[1] = *(const u32x4*)(arow + k0 + 8);
#pragma unroll
        for (int j = 0; j < 16; ++j)
            bm.u[j] = f32_to_bf16(bcol[(size_t)(k0 + j + kbof) * LORA_RANK]);
        c = __builtin_amdgcn_wmma_f32_16x16x32_bf16(false, a.v, false, bm.v,
                                                    (short)0, c, false, false);
    }
    float* out = ctx + ((size_t)b * NUM_HEADS + m0 + ((lane >> 4) << 3)) * LORA_RANK + n0 + l15;
#pragma unroll
    for (int i = 0; i < 8; ++i) out[(size_t)i * LORA_RANK] = c[i];
}

// ---------------------------------------------------------------------------
// K6: out[b,h,d] = Wv_h[d,:] . ctx[b,h,:] + bv_h[d]
// ---------------------------------------------------------------------------
__global__ void k6_outproj(const float* __restrict__ ctx, const float* __restrict__ kvb_w,
                           const float* __restrict__ kvb_b, float* __restrict__ out) {
    int idx = blockIdx.x * blockDim.x + threadIdx.x;   // (h, d)
    int d = idx & (HEAD_DIM - 1);
    int h = idx >> 7;
    if (h >= NUM_HEADS) return;
    int o = h * 256 + HEAD_DIM + d;
    const float* wrow = kvb_w + (size_t)o * LORA_RANK;
    float acc[BATCH];
#pragma unroll
    for (int b = 0; b < BATCH; ++b) acc[b] = 0.f;
#pragma unroll 4
    for (int r = 0; r < LORA_RANK; ++r) {
        float w = wrow[r];
#pragma unroll
        for (int b = 0; b < BATCH; ++b)
            acc[b] += ctx[((size_t)b * NUM_HEADS + h) * LORA_RANK + r] * w;
    }
    float bias = kvb_b[o];
#pragma unroll
    for (int b = 0; b < BATCH; ++b)
        out[(size_t)b * NUM_HEADS * HEAD_DIM + h * HEAD_DIM + d] = acc[b] + bias;
}

// ---------------------------------------------------------------------------
extern "C" void kernel_launch(void* const* d_in, const int* in_sizes, int n_in,
                              void* d_out, int out_size, void* d_ws, size_t ws_size,
                              hipStream_t stream) {
    const float* hidden = (const float*)d_in[0];
    const float* ckv    = (const float*)d_in[1];
    const float* wq     = (const float*)d_in[2];
    const float* bq     = (const float*)d_in[3];
    const float* kvbw   = (const float*)d_in[4];
    const float* kvbb   = (const float*)d_in[5];
    float* out = (float*)d_out;

    char* ws = (char*)d_ws;
    float*          q      = (float*)(ws + Q_OFF);
    unsigned short* qt     = (unsigned short*)(ws + QT_OFF);
    float*          scores = (float*)(ws + SC_OFF);
    unsigned short* probs  = (unsigned short*)(ws + PR_OFF);
    float*          ctx    = (float*)(ws + CTX_OFF);
    unsigned short* ckv16  = (unsigned short*)(ws + CKV16_OFF);
    unsigned short* ckvT   = (unsigned short*)(ws + CKVT_OFF);

    // Path choice is a pure function of ws_size: deterministic across calls.
    const int path = (ws_size >= WS_NEED_TR) ? 2 : (ws_size >= WS_NEED_PRE) ? 1 : 0;

    k1_qproj  <<<dim3(D_MODEL / 256), 256, 0, stream>>>(hidden, wq, bq, q);
    k2_qabsorb<<<dim3((NUM_HEADS * LORA_RANK) / 256), 256, 0, stream>>>(q, kvbw, qt);

    if (path == 2) {
        k0_cvt_tr<<<dim3(KV_LEN / 32, LORA_RANK / 32, BATCH), 256, 0, stream>>>(ckv, ckv16, ckvT);
        k3_scores_pre<<<dim3(KV_LEN / 64, 1, BATCH), 256, 0, stream>>>(qt, ckv16, scores);
    } else if (path == 1) {
        k0_cvt<<<dim3((BATCH * KV_LEN * LORA_RANK / 4) / 256), 256, 0, stream>>>(
            ckv, (unsigned int*)ckv16);
        k3_scores_pre<<<dim3(KV_LEN / 64, 1, BATCH), 256, 0, stream>>>(qt, ckv16, scores);
    } else {
        k3_scores_cvt<<<dim3(KV_LEN / 64, 1, BATCH), 256, 0, stream>>>(qt, ckv, scores);
    }

    k4_softmax<<<dim3(BATCH * NUM_HEADS), 256, 0, stream>>>(scores, probs);

    if (path == 2) {
        k5_ctx_tr<<<dim3(LORA_RANK / 64, 1, BATCH), 256, 0, stream>>>(probs, ckvT, ctx);
    } else if (path == 1) {
        k5_ctx_pre<<<dim3(LORA_RANK / 64, 1, BATCH), 256, 0, stream>>>(probs, ckv16, ctx);
    } else {
        k5_ctx_cvt<<<dim3(LORA_RANK / 64, 1, BATCH), 256, 0, stream>>>(probs, ckv, ctx);
    }

    k6_outproj<<<dim3((NUM_HEADS * HEAD_DIM) / 256), 256, 0, stream>>>(ctx, kvbw, kvbb, out);
}
